// ModulatedConv2d_59244778881225
// MI455X (gfx1250) — matrix operand order repaired and checked by reading, not compiled, not run on hardware
//
#include <hip/hip_runtime.h>

typedef unsigned short u16;
typedef __attribute__((ext_vector_type(16))) __bf16 v16bf;
typedef __attribute__((ext_vector_type(8)))  float  v8f;

union ABfrag { uint4 q[2]; v16bf v; };

__device__ __forceinline__ u16 f2bf(float f) {
    union { float f; unsigned u; } x; x.f = f;
    unsigned u = x.u;
    u += 0x7FFFu + ((u >> 16) & 1u);      // round-to-nearest-even
    return (u16)(u >> 16);
}

// ---------------- kernel 1: style = w @ affine_w^T + affine_b + 1 ----------
__global__ void style_kernel(const float* __restrict__ w,
                             const float* __restrict__ aw,
                             const float* __restrict__ ab,
                             float* __restrict__ s) {
    int idx = blockIdx.x * blockDim.x + threadIdx.x;   // 0..4095
    int b = idx >> 9, ci = idx & 511;
    const float* wr = w + b * 512;
    const float* ar = aw + ci * 512;
    float acc = 0.f;
#pragma unroll 4
    for (int d = 0; d < 512; d += 4) {
        float4 wv = *(const float4*)(wr + d);
        float4 av = *(const float4*)(ar + d);
        acc += wv.x * av.x + wv.y * av.y + wv.z * av.z + wv.w * av.w;
    }
    s[idx] = acc + ab[ci] + 1.0f;
}

// ------- kernel 2: modulate + demodulate + pack bf16 in WMMA-A order -------
// packed layout: [((b*9+kpos)*16 + cc)*16384 + co*32 + ci_in]   (u16)
__global__ void modpack_kernel(const float* __restrict__ weight,
                               const float* __restrict__ s,
                               u16* __restrict__ packed) {
    int gw   = (blockIdx.x * blockDim.x + threadIdx.x) >> 5;  // global wave 0..4095
    int lane = threadIdx.x & 31;
    int b  = gw >> 9;
    int co = gw & 511;
    const float* wbase = weight + (size_t)co * 512 * 9;   // [cin][3][3]
    const float* sb    = s + b * 512;

    float sum = 0.f;
    for (int i = lane; i < 4608; i += 32) {
        int ci = i / 9, kp = i - ci * 9;
        float v = wbase[ci * 9 + kp] * sb[ci];
        sum += v * v;
    }
#pragma unroll
    for (int off = 16; off; off >>= 1) sum += __shfl_xor(sum, off, 32);
    float dmod = rsqrtf(sum + 1e-8f);

    for (int i = lane; i < 4608; i += 32) {
        int ci = i / 9, kp = i - ci * 9;
        float v = wbase[ci * 9 + kp] * sb[ci] * dmod;
        int cc = ci >> 5, cil = ci & 31;
        size_t o = (((size_t)b * 9 + kp) * 16 + cc) * 16384 + (size_t)co * 32 + cil;
        packed[o] = f2bf(v);
    }
}

// ---------------- kernel 3: implicit-GEMM grouped conv via WMMA ------------
// grid: (co_half=2, row=64, batch=8); block: 256 (8 wave32)
// WG tile: M=256 cout x N=64 px (one image row); K = 512 cin x 9 taps.
// Waves split M 8-ways (32 cout each) and cover all 64 px: every A fragment
// feeds 4 WMMAs and each 16KB A chunk is read exactly once per workgroup.
// A fragments stream from global (fragment-order pack, L2-resident) with a
// TWO-chunk-deep register pipeline; input row staged in LDS as bf16.
// No barriers in the inner K loop. waves_per_eu(2) trades occupancy for
// in-flight loads.
#define B_STRIDE 264   /* u16 per px slot (256 ci + 8 pad); 528 B, 16B aligned */

__global__ __launch_bounds__(256)
__attribute__((amdgpu_waves_per_eu(2)))
void conv_kernel(const float* __restrict__ x,
                 const float* __restrict__ noise,
                 const float* __restrict__ bias,
                 const u16*   __restrict__ packedA,
                 float* __restrict__ out) {
    __shared__ alignas(16) u16 ldsB[66 * B_STRIDE];   // 34,848 B : [px slot][ci half]

    const int t    = threadIdx.x;
    const int lane = t & 31;
    const int wid  = t >> 5;             // wave 0..7 -> 32-cout slice
    const int l    = lane & 15;
    const int hi   = lane >> 4;          // K-half selector per WMMA 16-bit layout
    const int ct   = blockIdx.x;         // cout half (256 each)
    const int h    = blockIdx.y;         // output row
    const int b    = blockIdx.z;         // batch

    const int aoff = hi * 8;             // u16 offset: K-half select

    v8f c[2][4];                         // 32 cout x 64 px per wave
#pragma unroll
    for (int i = 0; i < 2; ++i)
#pragma unroll
        for (int j = 0; j < 4; ++j) c[i][j] = (v8f){};

    const size_t xb = (size_t)b * 512 * 64 * 64;

    for (int ky = 0; ky < 3; ++ky) {
        const int r    = h + ky - 1;
        const bool rin = (0 <= r) && (r < 64);
        for (int hh = 0; hh < 2; ++hh) {         // ci halves of 256
            // base of this stage's A chunks: q = kx*8 + cc8 -> 16KB chunk each
            const u16* Astage = packedA
                + (((size_t)b * 9 + ky * 3) * 16 + hh * 8) * 16384
                + (size_t)ct * 8192;
            __builtin_prefetch((const char*)Astage, 0, 3);

            __syncthreads();                     // prior B reads complete
            // stage one padded input row (66 slots x 256 ci) as bf16
            for (int e = t; e < 66 * 256; e += 256) {
                int ci   = e / 66;               // 0..255 within half
                int sidx = e - ci * 66;          // px slot 0..65 -> col sidx-1
                int cgl  = sidx - 1;
                float v = 0.f;
                if (rin && (unsigned)cgl < 64u)
                    v = x[xb + (((size_t)(hh * 256 + ci)) * 64 + r) * 64 + cgl];
                ldsB[sidx * B_STRIDE + ci] = f2bf(v);
            }
            __syncthreads();                     // B staged

            // chunk pointer: q in [0,24); kx = q>>3 steps 16 chunks, cc8 = q&7
            auto chunk = [&](int q) -> const u16* {
                int kx = q >> 3, cc8 = q & 7;
                return Astage + ((size_t)kx * 16 + cc8) * 16384;
            };
            auto loadA = [&](ABfrag (&dst)[2], const u16* ch) {
#pragma unroll
                for (int i = 0; i < 2; ++i) {
                    int co = wid * 32 + i * 16 + l;
                    const uint4* p = (const uint4*)(ch + co * 32 + aoff);
                    dst[i].q[0] = p[0];
                    dst[i].q[1] = p[2];          // +32 bytes
                }
            };

            ABfrag a[2], an[2], an2[2];
            loadA(a,  chunk(0));                 // 2-deep pipeline prologue
            loadA(an, chunk(1));

#pragma unroll
            for (int q = 0; q < 24; ++q) {
                const int kx  = q >> 3;
                const int cc8 = q & 7;

                // issue chunk q+2's A loads before chunk q's WMMAs:
                // guaranteed >=16-WMMA issue distance per load
                if (q + 2 < 24) loadA(an2, chunk(q + 2));

                ABfrag bb[4];
#pragma unroll
                for (int j = 0; j < 4; ++j) {
                    int slot = j * 16 + l + kx;             // col + kx - 1 + 1
                    const uint4* p = (const uint4*)
                        &ldsB[slot * B_STRIDE + cc8 * 32 + hi * 8];
                    bb[j].q[0] = p[0];           // K {8h..8h+7}, then {16+8h..16+8h+7}
                    bb[j].q[1] = p[2];
                }

#pragma unroll
                for (int i = 0; i < 2; ++i)
#pragma unroll
                    for (int j = 0; j < 4; ++j)
                        c[i][j] = __builtin_amdgcn_wmma_f32_16x16x32_bf16(
                            false, a[i].v, false, bb[j].v,
                            (short)0, c[i][j], false, false);

                // rotate pipeline registers (SSA under full unroll: no moves)
#pragma unroll
                for (int i = 0; i < 2; ++i) { a[i] = an[i]; an[i] = an2[i]; }
            }
        }
    }

    // epilogue: + noise + bias, leaky ReLU 0.2
#pragma unroll
    for (int i = 0; i < 2; ++i)
#pragma unroll
        for (int j = 0; j < 4; ++j) {
            int cobase = ct * 256 + wid * 32 + i * 16 + hi * 8;
            int col    = j * 16 + l;
#pragma unroll
            for (int r8 = 0; r8 < 8; ++r8) {
                int coo = cobase + r8;
                size_t o = (((size_t)b * 512 + coo) * 64 + h) * 64 + col;
                float y = c[i][j][r8] + noise[o] + bias[coo];
                out[o] = (y >= 0.f) ? y : 0.2f * y;
            }
        }
}

// --------------------------------- launch ----------------------------------
extern "C" void kernel_launch(void* const* d_in, const int* in_sizes, int n_in,
                              void* d_out, int out_size, void* d_ws, size_t ws_size,
                              hipStream_t stream) {
    const float* x        = (const float*)d_in[0];   // [8,512,64,64]
    const float* w        = (const float*)d_in[1];   // [8,512]
    const float* noise    = (const float*)d_in[2];   // [8,512,64,64]
    const float* weight   = (const float*)d_in[3];   // [1,512,512,3,3]
    const float* affine_w = (const float*)d_in[4];   // [512,512]
    const float* affine_b = (const float*)d_in[5];   // [512]
    const float* bias     = (const float*)d_in[6];   // [512]
    float* out = (float*)d_out;

    // workspace: [0,16KB) style s[8][512]; then packed bf16 weights (~37.75 MB)
    float* s_ws   = (float*)d_ws;
    u16*   packed = (u16*)((char*)d_ws + 16384);

    style_kernel<<<16, 256, 0, stream>>>(w, affine_w, affine_b, s_ws);
    modpack_kernel<<<512, 256, 0, stream>>>(weight, s_ws, packed);
    dim3 grid(2, 64, 8);
    conv_kernel<<<grid, 256, 0, stream>>>(x, noise, bias, packed, out);
}
